// AttentionLayer_28046136443210
// MI455X (gfx1250) — matrix operand order repaired and checked by reading, not compiled, not run on hardware
//
#include <hip/hip_runtime.h>
#include <cstdint>

#define BATCH 8
#define CH    512     // C == KC == OC
#define SEQ   4096    // N

typedef __bf16 bf16_t;
typedef bf16_t v16bf __attribute__((ext_vector_type(16)));
typedef float  v8f   __attribute__((ext_vector_type(8)));
typedef float  v4f   __attribute__((ext_vector_type(4)));
typedef unsigned short ushort8 __attribute__((ext_vector_type(8)));

union Frag16 {
  v16bf   v;
  ushort8 h[2];
};

__device__ __forceinline__ unsigned short f2bf(float f) {
  unsigned int u = __float_as_uint(f);
  u += 0x7FFFu + ((u >> 16) & 1u);   // round-to-nearest-even
  return (unsigned short)(u >> 16);
}

// ---------------------------------------------------------------------------
// x[b][c][n] fp32 -> xT[b][n][c] bf16 via LDS 64x64 tile transpose.
// Coalesced fp32 reads AND coalesced bf16 writes.
// grid = BATCH * (CH/64) * (SEQ/64) = 4096 blocks, 256 threads.
// ---------------------------------------------------------------------------
__global__ __launch_bounds__(256)
void cvt_x_kernel(const float* __restrict__ x, unsigned short* __restrict__ xT) {
  __shared__ unsigned short tile[64][72];   // padded rows to spread banks
  const int t     = threadIdx.x;
  const int ntile = blockIdx.x % (SEQ / 64);
  const int ctile = (blockIdx.x / (SEQ / 64)) % (CH / 64);
  const int b     = blockIdx.x / ((SEQ / 64) * (CH / 64));
  const int n0 = ntile * 64, c0 = ctile * 64;

  const float* xp = x + ((size_t)b * CH + c0) * SEQ + n0;
#pragma unroll
  for (int it = 0; it < 16; ++it) {
    int idx = t + it * 256;
    int i = idx >> 6;            // c within tile
    int j = idx & 63;            // n within tile (fastest -> coalesced read)
    tile[i][j] = f2bf(xp[(size_t)i * SEQ + j]);
  }
  __syncthreads();
  unsigned short* op = xT + ((size_t)b * SEQ + n0) * CH + c0;
#pragma unroll
  for (int it = 0; it < 16; ++it) {
    int idx = t + it * 256;
    int j = idx >> 6;            // n within tile
    int i = idx & 63;            // c within tile (fastest -> coalesced write)
    op[(size_t)j * CH + i] = tile[i][j];
  }
}

__global__ void cvt_w_kernel(const float* __restrict__ w,
                             unsigned short* __restrict__ o, int nelem) {
  int idx = blockIdx.x * blockDim.x + threadIdx.x;
  if (idx < nelem) o[idx] = f2bf(w[idx]);
}

// ---------------------------------------------------------------------------
// Projection GEMM:  Yt[n][k] = sum_c xT[b][n][c] * W[k][c]
// Double-buffered K-loop (loads for t+1 issued before WMMAs for t).
// transposed=1 -> Yt[b][n][k] via LDS-staged coalesced stores (Q,K)
// transposed=0 -> Y[b][k][n] via packed b128 stores (V)
// ---------------------------------------------------------------------------
__global__ __launch_bounds__(256)
void proj_kernel(const unsigned short* __restrict__ xT,
                 const unsigned short* __restrict__ Wb,
                 unsigned short* __restrict__ dst, int transposed) {
  __shared__ unsigned short st[8][16][64];
  const int lane = threadIdx.x & 31;
  const int w    = threadIdx.x >> 5;
  const int col  = lane & 15;
  const int hi   = lane >> 4;
  const int b    = blockIdx.x / (SEQ / 16);
  const int n0   = (blockIdx.x % (SEQ / 16)) * 16;
  const int k0   = w * 64;

  const unsigned short* ap = xT + ((size_t)b * SEQ + n0 + col) * CH + hi * 16;
  const unsigned short* bp[4];
#pragma unroll
  for (int ft = 0; ft < 4; ++ft)
    bp[ft] = Wb + (size_t)(k0 + ft * 16 + col) * CH + hi * 16;

  v8f acc[4] = {};
  Frag16 a_cur, b_cur[4], a_nxt, b_nxt[4];
  a_cur.h[0] = *(const ushort8*)(ap);
  a_cur.h[1] = *(const ushort8*)(ap + 8);
#pragma unroll
  for (int ft = 0; ft < 4; ++ft) {
    b_cur[ft].h[0] = *(const ushort8*)(bp[ft]);
    b_cur[ft].h[1] = *(const ushort8*)(bp[ft] + 8);
  }
#pragma unroll
  for (int t = 0; t < 16; ++t) {
    if (t + 1 < 16) {
      a_nxt.h[0] = *(const ushort8*)(ap + (t + 1) * 32);
      a_nxt.h[1] = *(const ushort8*)(ap + (t + 1) * 32 + 8);
#pragma unroll
      for (int ft = 0; ft < 4; ++ft) {
        b_nxt[ft].h[0] = *(const ushort8*)(bp[ft] + (t + 1) * 32);
        b_nxt[ft].h[1] = *(const ushort8*)(bp[ft] + (t + 1) * 32 + 8);
      }
    }
#pragma unroll
    for (int ft = 0; ft < 4; ++ft)
      acc[ft] = __builtin_amdgcn_wmma_f32_16x16x32_bf16(
          false, a_cur.v, false, b_cur[ft].v, (short)0, acc[ft], false, false);
    a_cur = a_nxt;
#pragma unroll
    for (int ft = 0; ft < 4; ++ft) b_cur[ft] = b_nxt[ft];
  }

  if (transposed) {
    // stage 16(n) x 64(k) bf16 tile in LDS, then coalesced 16B-chunk stores
#pragma unroll
    for (int ft = 0; ft < 4; ++ft)
#pragma unroll
      for (int v = 0; v < 8; ++v)
        st[w][v + hi * 8][ft * 16 + col] = f2bf(acc[ft][v]);
    __syncthreads();
#pragma unroll
    for (int cchunk = 0; cchunk < 2; ++cchunk) {
      int c   = lane + cchunk * 32;      // 0..63 chunk id
      int row = c >> 2;                  // n row 0..15
      int off = (c & 3) * 8;             // k offset 0..24
      ushort8 vdat = *(const ushort8*)&st[w][row][off];
      *(ushort8*)(dst + ((size_t)b * SEQ + n0 + row) * CH + k0 + off) = vdat;
    }
  } else {
    // V layout [k][n]: rows contiguous in n per lane -> packed b128 stores
#pragma unroll
    for (int ft = 0; ft < 4; ++ft) {
      ushort8 pack;
#pragma unroll
      for (int v = 0; v < 8; ++v) pack[v] = f2bf(acc[ft][v]);
      int k = k0 + ft * 16 + col;
      *(ushort8*)(dst + ((size_t)b * CH + k) * SEQ + n0 + hi * 8) = pack;
    }
  }
}

// ---------------------------------------------------------------------------
// Fused flash attention. Block = (b, 16-query tile). 8 waves.
// S phase: each wave does its own 16 keys of a 128-key block, double-buffered.
// Online softmax: shfl row-reductions + per-wave LDS slots + 2 barriers/block.
// PV phase: P (16x128 bf16, LDS) x V^T, double-buffered V fragments.
// ---------------------------------------------------------------------------
__global__ __launch_bounds__(256)
void attn_kernel(const unsigned short* __restrict__ Qt,
                 const unsigned short* __restrict__ Kt,
                 const unsigned short* __restrict__ V,
                 float* __restrict__ out) {
  __shared__ float          red_max[8][16];
  __shared__ float          red_sum[8][16];
  __shared__ unsigned short Ps[16][128];

  const int lane = threadIdx.x & 31;
  const int w    = threadIdx.x >> 5;
  const int col  = lane & 15;
  const int hi   = lane >> 4;
  const int b    = blockIdx.x / (SEQ / 16);
  const int n0   = (blockIdx.x % (SEQ / 16)) * 16;

  // Q A-fragments resident in VGPRs (16 tiles x 8 VGPRs = 128 VGPRs)
  Frag16 qf[16];
  {
    const unsigned short* qp = Qt + ((size_t)b * SEQ + n0 + col) * CH + hi * 16;
#pragma unroll
    for (int t = 0; t < 16; ++t) {
      qf[t].h[0] = *(const ushort8*)(qp + t * 32);
      qf[t].h[1] = *(const ushort8*)(qp + t * 32 + 8);
    }
  }

  v8f oacc[4] = {};
  float m_run[8], l_run[8];
#pragma unroll
  for (int v = 0; v < 8; ++v) { m_run[v] = -INFINITY; l_run[v] = 0.f; }

  const unsigned short* KtB = Kt + (size_t)b * SEQ * CH;
  const unsigned short* VB  = V  + (size_t)b * CH  * SEQ;

  for (int mb = 0; mb < SEQ / 128; ++mb) {
    const int mBase = mb * 128;

    // ---- S = Q(16xC) x K(C x 16) for this wave's 16 keys, double-buffered
    const unsigned short* kp =
        KtB + (size_t)(mBase + w * 16 + col) * CH + hi * 16;
    if (mb + 1 < SEQ / 128)
      __builtin_prefetch(kp + (size_t)128 * CH, 0, 0);
    v8f s = {};
    {
      Frag16 kc, kn;
      kc.h[0] = *(const ushort8*)(kp);
      kc.h[1] = *(const ushort8*)(kp + 8);
#pragma unroll
      for (int t = 0; t < 16; ++t) {
        if (t + 1 < 16) {
          kn.h[0] = *(const ushort8*)(kp + (t + 1) * 32);
          kn.h[1] = *(const ushort8*)(kp + (t + 1) * 32 + 8);
        }
        s = __builtin_amdgcn_wmma_f32_16x16x32_bf16(
            false, qf[t].v, false, kc.v, (short)0, s, false, false);
        kc = kn;
      }
    }

    // ---- row max over this wave's 16 keys (butterfly within 16-lane group)
    float rmax[8];
#pragma unroll
    for (int v = 0; v < 8; ++v) {
      float r = s[v];
      r = fmaxf(r, __shfl_xor(r, 8, 16));
      r = fmaxf(r, __shfl_xor(r, 4, 16));
      r = fmaxf(r, __shfl_xor(r, 2, 16));
      r = fmaxf(r, __shfl_xor(r, 1, 16));
      rmax[v] = r;
    }
    if (col == 0) {
#pragma unroll
      for (int v = 0; v < 8; ++v) red_max[w][v + hi * 8] = rmax[v];
    }
    __syncthreads();

    // ---- combine across waves; rescale; P = exp(S - m_new); P -> LDS
    float rsum[8];
#pragma unroll
    for (int v = 0; v < 8; ++v) {
      const int row = v + hi * 8;
      float mt = red_max[0][row];
#pragma unroll
      for (int ww = 1; ww < 8; ++ww) mt = fmaxf(mt, red_max[ww][row]);
      float m_new = fmaxf(m_run[v], mt);
      float scale = __expf(m_run[v] - m_new);
      m_run[v] = m_new;
      l_run[v] *= scale;
#pragma unroll
      for (int ft = 0; ft < 4; ++ft) oacc[ft][v] *= scale;
      float pv = __expf(s[v] - m_new);
      Ps[row][w * 16 + col] = f2bf(pv);
      float r = pv;
      r += __shfl_xor(r, 8, 16);
      r += __shfl_xor(r, 4, 16);
      r += __shfl_xor(r, 2, 16);
      r += __shfl_xor(r, 1, 16);
      rsum[v] = r;
    }
    if (col == 0) {
#pragma unroll
      for (int v = 0; v < 8; ++v) red_sum[w][v + hi * 8] = rsum[v];
    }
    __syncthreads();

#pragma unroll
    for (int v = 0; v < 8; ++v) {
      const int row = v + hi * 8;
      float ssum = 0.f;
#pragma unroll
      for (int ww = 0; ww < 8; ++ww) ssum += red_sum[ww][row];
      l_run[v] += ssum;
    }

    // ---- O += P(16x128) x V^T(128 x 64 feats of this wave)
    Frag16 pf[4];
#pragma unroll
    for (int kb = 0; kb < 4; ++kb) {
      const unsigned short* pp = &Ps[col][kb * 32 + hi * 16];
      pf[kb].h[0] = *(const ushort8*)(pp);
      pf[kb].h[1] = *(const ushort8*)(pp + 8);
    }
    {
      const unsigned short* vbase =
          VB + (size_t)(w * 64 + col) * SEQ + mBase + hi * 16;
      Frag16 vc, vn;
      vc.h[0] = *(const ushort8*)(vbase);
      vc.h[1] = *(const ushort8*)(vbase + 8);
#pragma unroll
      for (int idx = 0; idx < 16; ++idx) {
        const int kb = idx >> 2, ft = idx & 3;
        if (idx + 1 < 16) {
          const int kb2 = (idx + 1) >> 2, ft2 = (idx + 1) & 3;
          const unsigned short* vp =
              vbase + (size_t)(ft2 * 16) * SEQ + kb2 * 32;
          vn.h[0] = *(const ushort8*)(vp);
          vn.h[1] = *(const ushort8*)(vp + 8);
        }
        oacc[ft] = __builtin_amdgcn_wmma_f32_16x16x32_bf16(
            false, pf[kb].v, false, vc.v, (short)0, oacc[ft], false, false);
        vc = vn;
      }
    }
  }

  // ---- normalize and store fp32 out[b][o][n] with packed b128 stores
#pragma unroll
  for (int ft = 0; ft < 4; ++ft) {
    const int o = w * 64 + ft * 16 + col;
    v4f lo, hh;
#pragma unroll
    for (int v = 0; v < 4; ++v) lo[v] = oacc[ft][v] / l_run[v];
#pragma unroll
    for (int v = 0; v < 4; ++v) hh[v] = oacc[ft][v + 4] / l_run[v + 4];
    float* po = out + ((size_t)b * CH + o) * SEQ + n0 + hi * 8;
    *(v4f*)(po)     = lo;
    *(v4f*)(po + 4) = hh;
  }
}

// ---------------------------------------------------------------------------
extern "C" void kernel_launch(void* const* d_in, const int* in_sizes, int n_in,
                              void* d_out, int out_size, void* d_ws, size_t ws_size,
                              hipStream_t stream) {
  const float* x  = (const float*)d_in[0];
  const float* WQ = (const float*)d_in[1];
  const float* WK = (const float*)d_in[2];
  const float* WV = (const float*)d_in[3];
  float* out = (float*)d_out;

  // workspace layout (bf16 elements): xT | wq | wk | wv | Qt | Kt | V
  unsigned short* ws = (unsigned short*)d_ws;
  const size_t XSZ = (size_t)BATCH * CH * SEQ;   // 16,777,216 elems
  const size_t WSZ = (size_t)CH * CH;            // 262,144 elems
  unsigned short* xT = ws;
  unsigned short* wq = xT + XSZ;
  unsigned short* wk = wq + WSZ;
  unsigned short* wv = wk + WSZ;
  unsigned short* Qt = wv + WSZ;
  unsigned short* Kt = Qt + XSZ;
  unsigned short* Vb = Kt + XSZ;
  (void)in_sizes; (void)n_in; (void)out_size; (void)ws_size;

  cvt_x_kernel<<<BATCH * (CH / 64) * (SEQ / 64), 256, 0, stream>>>(x, xT);
  cvt_w_kernel<<<(unsigned)(WSZ / 256), 256, 0, stream>>>(WQ, wq, (int)WSZ);
  cvt_w_kernel<<<(unsigned)(WSZ / 256), 256, 0, stream>>>(WK, wk, (int)WSZ);
  cvt_w_kernel<<<(unsigned)(WSZ / 256), 256, 0, stream>>>(WV, wv, (int)WSZ);

  dim3 g(BATCH * (SEQ / 16));   // 2048 workgroups
  proj_kernel<<<g, 256, 0, stream>>>(xT, wq, Qt, 1);   // Q transposed
  proj_kernel<<<g, 256, 0, stream>>>(xT, wk, Kt, 1);   // K transposed
  proj_kernel<<<g, 256, 0, stream>>>(xT, wv, Vb, 0);   // V natural [o][m]
  attn_kernel<<<g, 256, 0, stream>>>(Qt, Kt, Vb, out);
}
